// RNNDecoder_17248588661417
// MI455X (gfx1250) — matrix operand order repaired and checked by reading, not compiled, not run on hardware
//
#include <hip/hip_runtime.h>
#include <hip/hip_bf16.h>
#include <math.h>
#include <stdint.h>

// ---------------------------------------------------------------------------
// Types for CDNA5 WMMA (wave32): v_wmma_f32_16x16x32_bf16
// ---------------------------------------------------------------------------
typedef __bf16 bf16_t;
typedef __attribute__((ext_vector_type(16))) bf16_t v16bf;
typedef __attribute__((ext_vector_type(8)))  bf16_t v8bf;
typedef __attribute__((ext_vector_type(8)))  float  v8f;

#define HDIM  512
#define BATCH 16
#define TSTEP 128
#define G3    1536          // 3*H
#define VOC   32000
#define MROWS 2048          // B*T

// ---------------------------------------------------------------------------
// Fragment loaders (contiguous k = 16*(lane/16) + s layout, wave32). Pointers are pre-offset to
// the k=0 element of this fragment slice; row stride passed explicitly.
// A 16x32 bf16: lane<16 -> M=lane, K=0..15; lane>=16 -> K=16..31
// ---------------------------------------------------------------------------
__device__ __forceinline__ v16bf load_a_frag(const bf16_t* rowbase, int lane) {
    return *reinterpret_cast<const v16bf*>(rowbase + ((lane >> 4) << 4));
}

// B 32x16 bf16: lane gives column n; lanes 0..15 hold K=0..15, lanes 16..31 K=16..31.
// A column of B is a contiguous row of the [N][K] weight (or staged LDS tile).
__device__ __forceinline__ v16bf load_b_frag(const bf16_t* colbase, int lane) {
    return *reinterpret_cast<const v16bf*>(colbase + ((lane >> 4) << 4));
}

__device__ __forceinline__ v8f wmma_bf16(v16bf a, v16bf b, v8f c) {
    return __builtin_amdgcn_wmma_f32_16x16x32_bf16(false, a, false, b, (short)0, c, false, false);
}

// ---------------------------------------------------------------------------
// CDNA5 async global->LDS copy (ASYNCcnt path).  VDST VGPR carries the 32-bit
// LDS byte address (low 32 bits of the generic pointer to a __shared__ object).
// ---------------------------------------------------------------------------
__device__ __forceinline__ void cp_async_16B(const bf16_t* g, bf16_t* s) {
    uint32_t lds = (uint32_t)(uintptr_t)s;
    asm volatile("global_load_async_to_lds_b128 %0, %1, off"
                 :: "v"(lds), "v"(g) : "memory");
}
__device__ __forceinline__ void wait_async0() {
    asm volatile("s_wait_asynccnt 0x0" ::: "memory");
}

// ---------------------------------------------------------------------------
// Prep kernels
// ---------------------------------------------------------------------------
__global__ void k_cvt_bf16(const float* __restrict__ src, bf16_t* __restrict__ dst, int n) {
    int i = blockIdx.x * blockDim.x + threadIdx.x;
    if (i < n) dst[i] = (bf16_t)src[i];
}

// x_bf[r=t*16+b][k] = bf16(relu(emb[tok][k])), tok = (t==0) ? BOS(=1) : tgt[b][t-1]
__global__ void k_embed(const float* __restrict__ emb, const int* __restrict__ tgt,
                        bf16_t* __restrict__ xbf) {
    int id = blockIdx.x * blockDim.x + threadIdx.x;  // 2048*512 total
    int r  = id >> 9;
    int k  = id & 511;
    int t  = r >> 4;
    int b  = r & 15;
    int tok = (t == 0) ? 1 : tgt[b * TSTEP + t - 1];
    float v = emb[tok * HDIM + k];
    xbf[id] = (bf16_t)(v > 0.f ? v : 0.f);
}

__global__ void k_init_h(const float* __restrict__ eh, float* __restrict__ h,
                         bf16_t* __restrict__ hbf) {
    int i = blockIdx.x * blockDim.x + threadIdx.x;
    if (i < BATCH * HDIM) { float v = eh[i]; h[i] = v; hbf[i] = (bf16_t)v; }
}

// ---------------------------------------------------------------------------
// WMMA GEMM with async-LDS staging: out[M x N] = A[M x 512] * W[N x 512]^T + bias
// 128x128 tile per WG, 512 threads = 16 waves in 4(M) x 4(N) grid; each wave
// owns a 32x32 block = 4 f32 accumulators (2x2 WMMA tiles).
// K is consumed in 32-wide slices double-buffered in LDS via ASYNCcnt copies.
// grid = (M/128, N/128)
// ---------------------------------------------------------------------------
__global__ __launch_bounds__(512)
void k_gemm_bias(const bf16_t* __restrict__ A, const bf16_t* __restrict__ W,
                 const float* __restrict__ bias, float* __restrict__ out, int ldc) {
    __shared__ bf16_t As[2][128 * 32];
    __shared__ bf16_t Bs[2][128 * 32];

    int tid  = threadIdx.x;
    int lane = tid & 31;
    int w    = tid >> 5;
    int wm   = w >> 2, wn = w & 3;
    int row0 = blockIdx.x * 128;
    int col0 = blockIdx.y * 128;

    // staging assignment: 512 threads x 16B = one 128x32 bf16 tile per buffer
    int sr   = tid >> 2;            // 0..127 : tile row (A) / tile col (B)
    int sseg = (tid & 3) << 3;      // element offset 0,8,16,24 within 32-wide slice

    const bf16_t* ga = A + (size_t)(row0 + sr) * HDIM + sseg;
    const bf16_t* gb = W + (size_t)(col0 + sr) * HDIM + sseg;
    int sidx = sr * 32 + sseg;

    // prologue: stage K-slice 0 into buffer 0
    cp_async_16B(ga, &As[0][sidx]);
    cp_async_16B(gb, &Bs[0][sidx]);
    wait_async0();
    __syncthreads();

    v8f acc00 = {}, acc01 = {}, acc10 = {}, acc11 = {};
    const int NK = HDIM / 32;       // 16 slices

    #pragma unroll
    for (int it = 0; it < NK; ++it) {
        int cur = it & 1;
        if (it + 1 < NK) {          // stage next slice while computing this one
            int nxt = cur ^ 1;
            cp_async_16B(ga + (it + 1) * 32, &As[nxt][sidx]);
            cp_async_16B(gb + (it + 1) * 32, &Bs[nxt][sidx]);
        }
        const bf16_t* as = As[cur];
        const bf16_t* bs = Bs[cur];
        v16bf a0 = load_a_frag(as + (wm * 32 +      (lane & 15)) * 32, lane);
        v16bf a1 = load_a_frag(as + (wm * 32 + 16 + (lane & 15)) * 32, lane);
        v16bf b0 = load_b_frag(bs + (wn * 32 +      (lane & 15)) * 32, lane);
        v16bf b1 = load_b_frag(bs + (wn * 32 + 16 + (lane & 15)) * 32, lane);
        acc00 = wmma_bf16(a0, b0, acc00);
        acc01 = wmma_bf16(a0, b1, acc01);
        acc10 = wmma_bf16(a1, b0, acc10);
        acc11 = wmma_bf16(a1, b1, acc11);
        wait_async0();              // next slice landed in LDS
        __syncthreads();            // and everyone is done reading 'cur'
    }

    // epilogue: C/D layout -> VGPR p holds row p (+8 for lanes 16..31)
    int rb = row0 + wm * 32 + ((lane >> 4) << 3);
    int c0 = col0 + wn * 32 + (lane & 15);
    float bi0 = bias[c0], bi1 = bias[c0 + 16];
    #pragma unroll
    for (int p = 0; p < 8; ++p) {
        out[(size_t)(rb + p)      * ldc + c0]      = acc00[p] + bi0;
        out[(size_t)(rb + p)      * ldc + c0 + 16] = acc01[p] + bi1;
        out[(size_t)(rb + 16 + p) * ldc + c0]      = acc10[p] + bi0;
        out[(size_t)(rb + 16 + p) * ldc + c0 + 16] = acc11[p] + bi1;
    }
}

// ---------------------------------------------------------------------------
// One GRU step (teacher-forced). grid = 8 blocks x 128 threads (4 waves).
// M=16 is a single WMMA tile row: direct global fragment loads, 3 gate
// accumulators per wave. Kernel boundary = global sync; h ping-pongs.
// ---------------------------------------------------------------------------
__global__ __launch_bounds__(128)
void k_gru_step(const float* __restrict__ xproj,   // [T*B][1536], r = t*16+b
                const bf16_t* __restrict__ whh,    // [1536][512] bf16
                const float* __restrict__ bhh,     // [1536]
                const float* __restrict__ h_in,    // [16][512] f32
                const bf16_t* __restrict__ hbf_in, // [16][512] bf16
                float* __restrict__ h_out,
                bf16_t* __restrict__ hbf_out,
                bf16_t* __restrict__ grub,         // [B*T][512] bf16, row = b*128+t
                int t) {
    int lane = threadIdx.x & 31;
    int w    = threadIdx.x >> 5;
    int jw   = blockIdx.x * 64 + w * 16;

    const bf16_t* arow = hbf_in + (lane & 15) * HDIM;
    const bf16_t* wr   = whh + (size_t)(       jw + (lane & 15)) * HDIM;
    const bf16_t* wz   = whh + (size_t)( 512 + jw + (lane & 15)) * HDIM;
    const bf16_t* wn   = whh + (size_t)(1024 + jw + (lane & 15)) * HDIM;

    v8f ar = {}, az = {}, an = {};
    #pragma unroll
    for (int k0 = 0; k0 < HDIM; k0 += 32) {
        __builtin_prefetch(wr + k0 + 64, 0, 1);   // global_prefetch_b8
        v16bf a = load_a_frag(arow + k0, lane);
        ar = wmma_bf16(a, load_b_frag(wr + k0, lane), ar);
        az = wmma_bf16(a, load_b_frag(wz + k0, lane), az);
        an = wmma_bf16(a, load_b_frag(wn + k0, lane), an);
    }

    int j = jw + (lane & 15);
    int bbase = ((lane >> 4) << 3);
    float br = bhh[j], bz = bhh[512 + j], bn = bhh[1024 + j];
    #pragma unroll
    for (int p = 0; p < 8; ++p) {
        int b = bbase + p;
        const float* xp = xproj + (size_t)(t * 16 + b) * G3;
        float hr = ar[p] + br;
        float hz = az[p] + bz;
        float hn = an[p] + bn;
        float r  = 1.f / (1.f + __expf(-(xp[j] + hr)));
        float z  = 1.f / (1.f + __expf(-(xp[512 + j] + hz)));
        float n  = tanhf(xp[1024 + j] + r * hn);
        float hprev = h_in[b * HDIM + j];
        float hnew  = (1.f - z) * n + z * hprev;
        h_out[b * HDIM + j]   = hnew;
        hbf_out[b * HDIM + j] = (bf16_t)hnew;
        grub[(size_t)(b * TSTEP + t) * HDIM + j] = (bf16_t)hnew;
    }
}

// ---------------------------------------------------------------------------
// In-place log_softmax over V per row; one workgroup per row (rows stay in L2).
// ---------------------------------------------------------------------------
__global__ __launch_bounds__(256)
void k_logsoftmax(float* __restrict__ out) {
    __shared__ float red[256];
    float* x = out + (size_t)blockIdx.x * VOC;
    int tid = threadIdx.x;

    float m = -3.402823466e38f;
    for (int i = tid; i < VOC; i += 256) m = fmaxf(m, x[i]);
    red[tid] = m; __syncthreads();
    for (int s = 128; s > 0; s >>= 1) {
        if (tid < s) red[tid] = fmaxf(red[tid], red[tid + s]);
        __syncthreads();
    }
    m = red[0]; __syncthreads();

    float sum = 0.f;
    for (int i = tid; i < VOC; i += 256) sum += __expf(x[i] - m);
    red[tid] = sum; __syncthreads();
    for (int s = 128; s > 0; s >>= 1) {
        if (tid < s) red[tid] += red[tid + s];
        __syncthreads();
    }
    float L = m + logf(red[0]);

    for (int i = tid; i < VOC; i += 256) x[i] = x[i] - L;
}

__global__ void k_copy(const float* __restrict__ src, float* __restrict__ dst, int n) {
    int i = blockIdx.x * blockDim.x + threadIdx.x;
    if (i < n) dst[i] = src[i];
}

// ---------------------------------------------------------------------------
// Host launcher
// ---------------------------------------------------------------------------
extern "C" void kernel_launch(void* const* d_in, const int* in_sizes, int n_in,
                              void* d_out, int out_size, void* d_ws, size_t ws_size,
                              hipStream_t stream) {
    (void)in_sizes; (void)n_in; (void)ws_size;
    const float* enc_hidden = (const float*)d_in[1];   // [1,16,512]
    const int*   tgt        = (const int*)  d_in[2];   // [16,128]
    const float* emb        = (const float*)d_in[3];   // [32000,512]
    const float* w_ih       = (const float*)d_in[4];   // [1536,512]
    const float* b_ih       = (const float*)d_in[5];
    const float* w_hh       = (const float*)d_in[6];   // [1536,512]
    const float* b_hh       = (const float*)d_in[7];
    const float* w_out      = (const float*)d_in[8];   // [32000,512]
    const float* b_out      = (const float*)d_in[9];

    // workspace carve-up (256B aligned)
    char* ws = (char*)d_ws;
    size_t off = 0;
    auto carve = [&](size_t bytes) {
        char* p = ws + off;
        off = (off + bytes + 255) & ~(size_t)255;
        return p;
    };
    float*  xproj   = (float*) carve((size_t)MROWS * G3 * 4);      // 12.6 MB
    bf16_t* grub    = (bf16_t*)carve((size_t)MROWS * HDIM * 2);    //  2.1 MB
    bf16_t* xbf     = (bf16_t*)carve((size_t)MROWS * HDIM * 2);    //  2.1 MB
    bf16_t* wih_bf  = (bf16_t*)carve((size_t)G3 * HDIM * 2);       //  1.6 MB
    bf16_t* whh_bf  = (bf16_t*)carve((size_t)G3 * HDIM * 2);       //  1.6 MB
    bf16_t* wout_bf = (bf16_t*)carve((size_t)VOC * HDIM * 2);      // 32.8 MB
    float*  hA      = (float*) carve((size_t)BATCH * HDIM * 4);
    float*  hB      = (float*) carve((size_t)BATCH * HDIM * 4);
    bf16_t* hbA     = (bf16_t*)carve((size_t)BATCH * HDIM * 2);
    bf16_t* hbB     = (bf16_t*)carve((size_t)BATCH * HDIM * 2);

    // 1) weights -> bf16
    {
        int n1 = G3 * HDIM;
        k_cvt_bf16<<<(n1 + 255) / 256, 256, 0, stream>>>(w_ih, wih_bf, n1);
        k_cvt_bf16<<<(n1 + 255) / 256, 256, 0, stream>>>(w_hh, whh_bf, n1);
        int n2 = VOC * HDIM;
        k_cvt_bf16<<<(n2 + 255) / 256, 256, 0, stream>>>(w_out, wout_bf, n2);
    }
    // 2) embedding gather + ReLU -> bf16
    k_embed<<<(MROWS * HDIM) / 256, 256, 0, stream>>>(emb, tgt, xbf);
    // 3) h0 = encoder_hidden
    k_init_h<<<(BATCH * HDIM + 255) / 256, 256, 0, stream>>>(enc_hidden, hA, hbA);

    // 4) x_proj = relu(emb[...]) @ w_ih^T + b_ih   (WMMA + async-LDS)
    k_gemm_bias<<<dim3(MROWS / 128, G3 / 128), 512, 0, stream>>>(xbf, wih_bf, b_ih, xproj, G3);

    // 5) sequential GRU scan, kernel per step (launch boundary = global sync)
    float*  hbuf[2]  = {hA, hB};
    bf16_t* hbbuf[2] = {hbA, hbB};
    for (int t = 0; t < TSTEP; ++t) {
        int i = t & 1;
        k_gru_step<<<dim3(8), 128, 0, stream>>>(xproj, whh_bf, b_hh,
                                                hbuf[i], hbbuf[i],
                                                hbuf[1 - i], hbbuf[1 - i],
                                                grub, t);
    }
    // after t=127 (in = buf1, out = buf0), final hidden lives in hbuf[0]

    // 6) logits = gru_out @ w_out^T + b_out -> straight into d_out (WMMA + async-LDS)
    k_gemm_bias<<<dim3(MROWS / 128, VOC / 128), 512, 0, stream>>>(grub, wout_bf, b_out,
                                                                  (float*)d_out, VOC);
    // 7) in-place log_softmax per row
    k_logsoftmax<<<MROWS, 256, 0, stream>>>((float*)d_out);

    // 8) hidden -> tail of d_out
    float* hid_out = (float*)d_out + ((size_t)out_size - BATCH * HDIM);
    k_copy<<<(BATCH * HDIM + 255) / 256, 256, 0, stream>>>(hbuf[0], hid_out, BATCH * HDIM);
}
